// Encoder_75333726371971
// MI455X (gfx1250) — matrix operand order repaired and checked by reading, not compiled, not run on hardware
//
#include <hip/hip_runtime.h>
#include <hip/hip_bf16.h>

// ---------------- problem constants (match reference) ----------------
#define NN   50000
#define EE   800000
#define DIN  512
#define DH1  256
#define DH2  128
#define NEG_SLOPE 0.2f

// ---------------- WMMA vector types (CDNA5 wave32) ----------------
typedef __attribute__((ext_vector_type(16))) __bf16 v16bf;
typedef __attribute__((ext_vector_type(8)))  __bf16 v8bf;
typedef __attribute__((ext_vector_type(8)))  float  v8f;

__device__ __forceinline__ __bf16 to_bf16(float f) {
    __hip_bfloat16 h = __float2bfloat16(f);
    __bf16 b;
    __builtin_memcpy(&b, &h, 2);
    return b;
}

__device__ __forceinline__ float leaky(float v) {
    return v > 0.0f ? v : NEG_SLOPE * v;
}

// ---------------- utility kernels ----------------
__global__ void fill_zero_kernel(float* __restrict__ p, long n) {
    long t = (long)blockIdx.x * blockDim.x + threadIdx.x;
    if (t < n) p[t] = 0.0f;
}

// degree accumulation: deg[dst] += 1 per edge
__global__ void degree_kernel(const long long* __restrict__ ei,
                              float* __restrict__ deg, int nE) {
    int e = blockIdx.x * blockDim.x + threadIdx.x;
    if (e < nE) {
        int d = (int)ei[nE + e];        // row 1 of (2,E) = dst
        atomicAdd(&deg[d], 1.0f);
    }
}

// dinv = rsqrt(deg + 1)  (self-loop), in place
__global__ void rsqrt_kernel(float* __restrict__ dinv, int n) {
    int t = blockIdx.x * blockDim.x + threadIdx.x;
    if (t < n) dinv[t] = rsqrtf(dinv[t] + 1.0f);
}

// fp32 -> bf16 elementwise
__global__ void cvt_bf16_kernel(const float* __restrict__ x,
                                __bf16* __restrict__ y, long n) {
    long t = (long)blockIdx.x * blockDim.x + threadIdx.x;
    if (t < n) y[t] = to_bf16(x[t]);
}

// Pack weight W[K][Ncols] (row major fp32) into bf16 B-operand layout:
// Wp[((k/32)*Ncols + n)*32 + (k%32)]  -> each column's 32 K-values contiguous.
__global__ void pack_w_kernel(const float* __restrict__ W,
                              __bf16* __restrict__ Wp, int K, int Ncols) {
    int t = blockIdx.x * blockDim.x + threadIdx.x;
    if (t < K * Ncols) {
        int k = t / Ncols;
        int n = t - k * Ncols;
        Wp[((size_t)(k >> 5) * Ncols + n) * 32 + (k & 31)] = to_bf16(W[t]);
    }
}

// ---------------- WMMA GEMM: C[M x NDIM] = A[M x KDIM] * B ----------------
// A: bf16 row-major. Bp: packed [KDIM/32][NDIM][32] bf16. C: fp32 row-major.
// One wave computes a 16x16 tile; 8 waves per block share the same 16 A rows.
template <int KDIM, int NDIM>
__global__ __launch_bounds__(256)
void gemm_bf16_wmma_kernel(const __bf16* __restrict__ A,
                           const __bf16* __restrict__ Bp,
                           float* __restrict__ C, int M) {
    const int lane = threadIdx.x & 31;
    const int wave = threadIdx.x >> 5;
    const int row0 = blockIdx.x << 4;
    const int col0 = ((blockIdx.y << 3) + wave) << 4;
    if (row0 >= M || col0 >= NDIM) return;      // wave-uniform: EXEC stays full

    const int mr   = lane & 15;                 // A row within tile / B column
    const int half = lane >> 4;

    // A 16-bit layout: VGPR0..3 <- K = k0 + half*8 + [0..7]
    //                  VGPR4..7 <- K = k0 + 16 + half*8 + [0..7]
    const __bf16* ap = A + (size_t)(row0 + mr) * KDIM + (half << 3);
    // B layout: lane holds K = k0 + half*8 + [0..7] and k0 + 16 + half*8 + [0..7] of column n.
    const __bf16* bp = Bp + (((size_t)col0 + mr) << 5) + (half << 3);

    v8f acc = {};
#pragma unroll
    for (int kt = 0; kt < KDIM / 32; ++kt) {
        v8bf alo = *reinterpret_cast<const v8bf*>(ap);
        v8bf ahi = *reinterpret_cast<const v8bf*>(ap + 16);
        v8bf blo = *reinterpret_cast<const v8bf*>(bp);
        v8bf bhi = *reinterpret_cast<const v8bf*>(bp + 16);
        if (kt + 1 < KDIM / 32) __builtin_prefetch(ap + 32, 0, 3);

        v16bf av, bv;
#pragma unroll
        for (int i = 0; i < 8; ++i) {
            av[i] = alo[i]; av[i + 8] = ahi[i];
            bv[i] = blo[i]; bv[i + 8] = bhi[i];
        }
        acc = __builtin_amdgcn_wmma_f32_16x16x32_bf16(
            false, av, false, bv, (short)0, acc, false, false);

        ap += 32;
        bp += (size_t)NDIM * 32;
    }

    // C layout: VGPR v holds row (v + 8*half), column = lane&15
    float* cp = C + (size_t)row0 * NDIM + col0 + mr;
#pragma unroll
    for (int v = 0; v < 8; ++v)
        cp[(size_t)(v + (half << 3)) * NDIM] = acc[v];
}

// ---------------- edge-parallel aggregation (scatter-add) ----------------
// thread -> (edge, 4-feature chunk); consecutive lanes = consecutive features
__global__ void aggregate_kernel(const long long* __restrict__ ei,
                                 const float* __restrict__ dinv,
                                 const float* __restrict__ H,
                                 float* __restrict__ agg,
                                 int nE, int F) {
    long tid = (long)blockIdx.x * blockDim.x + threadIdx.x;
    const int chunks = F >> 2;
    if (tid >= (long)nE * chunks) return;
    int e  = (int)(tid / chunks);
    int f4 = (int)(tid - (long)e * chunks) << 2;
    int s = (int)ei[e];
    int d = (int)ei[nE + e];
    float coef = dinv[s] * dinv[d];
    const float4 hv = *reinterpret_cast<const float4*>(H + (size_t)s * F + f4);
    float* ap = agg + (size_t)d * F + f4;
    atomicAdd(ap + 0, hv.x * coef);
    atomicAdd(ap + 1, hv.y * coef);
    atomicAdd(ap + 2, hv.z * coef);
    atomicAdd(ap + 3, hv.w * coef);
}

// ---------------- fused epilogues ----------------
// z = leaky(agg + dinv^2 * h + b) -> bf16 (feeds next GEMM only)
__global__ void post_bf16_kernel(const float* __restrict__ agg,
                                 const float* __restrict__ H,
                                 const float* __restrict__ dinv,
                                 const float* __restrict__ bias,
                                 __bf16* __restrict__ out, int F) {
    long tid = (long)blockIdx.x * blockDim.x + threadIdx.x;
    if (tid >= (long)NN * F) return;
    int node = (int)(tid / F);
    int f    = (int)(tid - (long)node * F);
    float di = dinv[node];
    out[tid] = to_bf16(leaky(agg[tid] + di * di * H[tid] + bias[f]));
}

// final layer: fp32 output
__global__ void post_f32_kernel(const float* __restrict__ agg,
                                const float* __restrict__ H,
                                const float* __restrict__ dinv,
                                const float* __restrict__ bias,
                                float* __restrict__ out, int F) {
    long tid = (long)blockIdx.x * blockDim.x + threadIdx.x;
    if (tid >= (long)NN * F) return;
    int node = (int)(tid / F);
    int f    = (int)(tid - (long)node * F);
    float di = dinv[node];
    out[tid] = leaky(agg[tid] + di * di * H[tid] + bias[f]);
}

// ---------------- host launcher ----------------
static inline int nblk(long n, int b) { return (int)((n + b - 1) / b); }
static inline size_t alignup(size_t x) { return (x + 255) & ~(size_t)255; }

extern "C" void kernel_launch(void* const* d_in, const int* in_sizes, int n_in,
                              void* d_out, int out_size, void* d_ws, size_t ws_size,
                              hipStream_t stream) {
    const float*     X  = (const float*)d_in[0];
    const long long* EI = (const long long*)d_in[1];   // int64 (2, E)
    const float*     W1 = (const float*)d_in[2];
    const float*     b1 = (const float*)d_in[3];
    const float*     W2 = (const float*)d_in[4];
    const float*     b2 = (const float*)d_in[5];
    float* OUT = (float*)d_out;

    // ---- workspace carve-out (~154 MB with reuse) ----
    char* ws = (char*)d_ws;
    size_t off = 0;
    float*  dinv = (float*)(ws + off);  off += alignup(sizeof(float) * NN);
    __bf16* Xbf  = (__bf16*)(ws + off);
    size_t  xbf_off = off;              off += alignup(sizeof(__bf16) * (size_t)NN * DIN);
    __bf16* W1p  = (__bf16*)(ws + off); off += alignup(sizeof(__bf16) * DIN * DH1);
    __bf16* W2p  = (__bf16*)(ws + off); off += alignup(sizeof(__bf16) * DH1 * DH2);
    float*  H1f  = (float*)(ws + off);  off += alignup(sizeof(float) * (size_t)NN * DH1);
    float*  big  = (float*)(ws + off);  off += alignup(sizeof(float) * (size_t)NN * DH1);
    float*  agg1 = big;                            // N x 256 (consumed by post1)
    __bf16* Z1bf = (__bf16*)(ws + xbf_off);        // reuse X-bf16 region after GEMM1
    float*  H2f  = big;                            // reuse agg1 region: N x 128
    float*  agg2 = big + (size_t)NN * DH2;         // second half: N x 128
    (void)ws_size; (void)in_sizes; (void)n_in; (void)out_size;

    const int T = 256;

    // 1) degree -> dinv
    fill_zero_kernel<<<nblk(NN, T), T, 0, stream>>>(dinv, NN);
    degree_kernel<<<nblk(EE, T), T, 0, stream>>>(EI, dinv, EE);
    rsqrt_kernel<<<nblk(NN, T), T, 0, stream>>>(dinv, NN);

    // 2) operand prep
    cvt_bf16_kernel<<<nblk((long)NN * DIN, T), T, 0, stream>>>(X, Xbf, (long)NN * DIN);
    pack_w_kernel<<<nblk(DIN * DH1, T), T, 0, stream>>>(W1, W1p, DIN, DH1);
    pack_w_kernel<<<nblk(DH1 * DH2, T), T, 0, stream>>>(W2, W2p, DH1, DH2);

    // 3) H1 = X @ W1   (WMMA bf16 -> fp32)
    gemm_bf16_wmma_kernel<DIN, DH1>
        <<<dim3(NN / 16, DH1 / 128), T, 0, stream>>>(Xbf, W1p, H1f, NN);

    // 4) layer-1 aggregation + epilogue -> Z1 (bf16)
    fill_zero_kernel<<<nblk((long)NN * DH1, T), T, 0, stream>>>(agg1, (long)NN * DH1);
    aggregate_kernel<<<nblk((long)EE * (DH1 / 4), T), T, 0, stream>>>(
        EI, dinv, H1f, agg1, EE, DH1);
    post_bf16_kernel<<<nblk((long)NN * DH1, T), T, 0, stream>>>(
        agg1, H1f, dinv, b1, Z1bf, DH1);

    // 5) H2 = Z1 @ W2  (WMMA)
    gemm_bf16_wmma_kernel<DH1, DH2>
        <<<dim3(NN / 16, DH2 / 128), T, 0, stream>>>(Z1bf, W2p, H2f, NN);

    // 6) layer-2 aggregation + epilogue -> out (fp32)
    fill_zero_kernel<<<nblk((long)NN * DH2, T), T, 0, stream>>>(agg2, (long)NN * DH2);
    aggregate_kernel<<<nblk((long)EE * (DH2 / 4), T), T, 0, stream>>>(
        EI, dinv, H2f, agg2, EE, DH2);
    post_f32_kernel<<<nblk((long)NN * DH2, T), T, 0, stream>>>(
        agg2, H2f, dinv, b2, OUT, DH2);
}
